// VectorQuantizer_54511724921598
// MI455X (gfx1250) — matrix operand order repaired and checked by reading, not compiled, not run on hardware
//
#include <hip/hip_runtime.h>
#include <hip/hip_bf16.h>

// ---- CDNA5 WMMA vector types ----
typedef __bf16        v16bf __attribute__((ext_vector_type(16)));
typedef float         v8f   __attribute__((ext_vector_type(8)));
typedef unsigned int  v4u   __attribute__((ext_vector_type(4)));

#define KCODES 1024
#define DDIM   256
#define NROWS  65536            // 64 * 32 * 32
#define OUT_ELEMS 4194304       // 64 * 256 * 32 * 32
#define APAD   264              // padded LDS row stride (ushorts) -> bank-conflict free
#define GATHER_BLOCKS 2048      // NROWS / 32

__device__ __forceinline__ unsigned short f32_to_bf16(float f) {
    unsigned int u = __float_as_uint(f);
    unsigned int r = u + 0x7FFFu + ((u >> 16) & 1u);   // round-to-nearest-even
    return (unsigned short)(r >> 16);
}

// ---------------------------------------------------------------------------
// Kernel 1: codebook fp32 -> bf16, and e_norm[k] = sum_d e[k,d]^2
// grid: 1024 blocks x 256 threads (one block per code row)
// ---------------------------------------------------------------------------
__global__ __launch_bounds__(256) void vq_prep_emb(const float* __restrict__ emb,
                                                   unsigned short* __restrict__ eb,
                                                   float* __restrict__ enorm) {
    __shared__ float shR[256];
    const int k = blockIdx.x;
    const int t = threadIdx.x;
    float e = emb[k * DDIM + t];
    eb[k * DDIM + t] = f32_to_bf16(e);
    shR[t] = e * e;
    __syncthreads();
    for (int s = 128; s > 0; s >>= 1) {
        if (t < s) shR[t] += shR[t + s];
        __syncthreads();
    }
    if (t == 0) enorm[k] = shR[0];
}

// ---------------------------------------------------------------------------
// Kernel 2: argmin over codes via bf16 WMMA.  score = ||e||^2 - 2 * x.e
// grid: 512 blocks x 128 threads (4 waves); each wave owns 32 latent rows
// (two 16x16 M tiles -> two independent WMMA accumulator chains, A resident
// in VGPRs across all 64 code tiles).
// ---------------------------------------------------------------------------
__global__ __launch_bounds__(128) void vq_argmin(const float* __restrict__ lat,
                                                 const unsigned short* __restrict__ eb,
                                                 const float* __restrict__ enorm,
                                                 int* __restrict__ inds) {
    __shared__ __align__(16) unsigned short shAst[4][16 * APAD];  // per-wave staging
    __shared__ __align__(16) unsigned short shE[2][16 * APAD];    // dbl-buffered code tile
    __shared__ float shEn[2][16];

    const int tid  = threadIdx.x;
    const int wave = tid >> 5;
    const int lane = tid & 31;
    const int n0   = blockIdx.x * 128 + wave * 32;  // first latent row of this wave
    const int b    = n0 >> 10;                      // batch index (HW = 1024)
    const int hw0  = n0 & 1023;

    const int m  = lane & 15;   // A row / B column / C column within 16x16 tile
    const int hi = lane >> 4;   // lane-half selector per ISA VGPR layouts

    // cooperative loader for one 16-code tile (8KB bf16 + 16 norms)
    auto load_etile = [&](int kt2, int buf2) {
        const int c0  = kt2 << 4;
        const int row = tid >> 3;       // 16 rows, 8 threads per row
        const int seg = tid & 7;        // 8 x 64B segments per row
        const v4u* src = (const v4u*)(eb + (c0 + row) * DDIM + seg * 32);
        v4u* dst = (v4u*)(&shE[buf2][row * APAD + seg * 32]);
        dst[0] = src[0]; dst[1] = src[1]; dst[2] = src[2]; dst[3] = src[3];
        if (tid < 16) shEn[buf2][tid] = enorm[c0 + tid];
    };

    // ---- build register-resident A fragments for the wave's two M tiles ----
    v16bf afrag[2][8];
#pragma unroll
    for (int T = 0; T < 2; ++T) {
        // stage 16 rows x 256 d via transpose-gather (row pairs contiguous in HW)
        {
            const int r2   = (lane & 7) * 2;   // row pair
            const int dsel = lane >> 3;        // 0..3
            const int hwT  = hw0 + T * 16;
            const float* src = lat + (b * DDIM) * 1024 + hwT + r2;
            for (int d0 = 0; d0 < DDIM; d0 += 4) {
                int d = d0 + dsel;
                float2 x = *(const float2*)(src + d * 1024);
                shAst[wave][r2 * APAD + d]       = f32_to_bf16(x.x);
                shAst[wave][(r2 + 1) * APAD + d] = f32_to_bf16(x.y);
            }
        }
        __syncthreads();
        // gather fragments: lanes 0-15 -> d {dk+0..7, dk+16..23}; 16-31 -> {+8..15,+24..31}
#pragma unroll
        for (int dk8 = 0; dk8 < 8; ++dk8) {
            const unsigned short* ap = &shAst[wave][m * APAD + dk8 * 32 + hi * 8];
            v16bf av;
            ((v4u*)&av)[0] = *(const v4u*)(ap);
            ((v4u*)&av)[1] = *(const v4u*)(ap + 16);
            afrag[T][dk8] = av;
        }
        __syncthreads();
    }

    load_etile(0, 0);
    __syncthreads();

    float minv0[8], minv1[8];
    int   mini0[8], mini1[8];
#pragma unroll
    for (int r = 0; r < 8; ++r) {
        minv0[r] = 3.0e38f; mini0[r] = 0;
        minv1[r] = 3.0e38f; mini1[r] = 0;
    }

    for (int kt = 0; kt < KCODES / 16; ++kt) {
        const int buf = kt & 1;
        if (kt + 1 < KCODES / 16) load_etile(kt + 1, buf ^ 1);

        const float en = shEn[buf][m];
        const unsigned short* Bb = &shE[buf][0];

        // two independent 16x16 score tiles sharing each B fragment
        v8f acc0 = {0.f, 0.f, 0.f, 0.f, 0.f, 0.f, 0.f, 0.f};
        v8f acc1 = {0.f, 0.f, 0.f, 0.f, 0.f, 0.f, 0.f, 0.f};
#pragma unroll
        for (int dk8 = 0; dk8 < 8; ++dk8) {
            // B fragment: lanes 0-15 -> d {dk+0..7, dk+16..23}; 16-31 -> {+8..15,+24..31}
            const unsigned short* bp = Bb + m * APAD + dk8 * 32 + hi * 8;
            v16bf bv;
            ((v4u*)&bv)[0] = *(const v4u*)(bp);
            ((v4u*)&bv)[1] = *(const v4u*)(bp + 16);
            acc0 = __builtin_amdgcn_wmma_f32_16x16x32_bf16(
                false, afrag[0][dk8], false, bv, (short)0, acc0, false, false);
            acc1 = __builtin_amdgcn_wmma_f32_16x16x32_bf16(
                false, afrag[1][dk8], false, bv, (short)0, acc1, false, false);
        }

        // ---- running argmin update: score = ||e||^2 - 2*dot ----
        const int c = (kt << 4) + m;
#pragma unroll
        for (int r = 0; r < 8; ++r) {
            float s0 = __builtin_fmaf(-2.0f, acc0[r], en);
            if (s0 < minv0[r]) { minv0[r] = s0; mini0[r] = c; }
            float s1 = __builtin_fmaf(-2.0f, acc1[r], en);
            if (s1 < minv1[r]) { minv1[r] = s1; mini1[r] = c; }
        }
        __syncthreads();
    }

    // ---- butterfly reduce across the 16 lanes holding each row group ----
#pragma unroll
    for (int r = 0; r < 8; ++r) {
        for (int off = 1; off < 16; off <<= 1) {
            float ov = __shfl_xor(minv0[r], off, 32);
            int   oi = __shfl_xor(mini0[r], off, 32);
            if (ov < minv0[r] || (ov == minv0[r] && oi < mini0[r])) {
                minv0[r] = ov; mini0[r] = oi;
            }
            float ow = __shfl_xor(minv1[r], off, 32);
            int   oj = __shfl_xor(mini1[r], off, 32);
            if (ow < minv1[r] || (ow == minv1[r] && oj < mini1[r])) {
                minv1[r] = ow; mini1[r] = oj;
            }
        }
    }
    if ((lane & 15) == 0) {
        // lane 0 -> rows 0..7, lane 16 -> rows 8..15  (C/D layout)
#pragma unroll
        for (int r = 0; r < 8; ++r) {
            inds[n0 + hi * 8 + r]      = mini0[r];
            inds[n0 + 16 + hi * 8 + r] = mini1[r];
        }
    }
}

// ---------------------------------------------------------------------------
// Kernel 3: gather codes -> output (back to [B,D,H,W]) + loss partial sums.
// grid: 2048 blocks x 256 threads (32 latent rows per block)
// ---------------------------------------------------------------------------
__global__ __launch_bounds__(256) void vq_gather_out(const float* __restrict__ lat,
                                                     const float* __restrict__ emb,
                                                     const int* __restrict__ inds,
                                                     float* __restrict__ outp,
                                                     float* __restrict__ partial) {
    __shared__ int   shI[32];
    __shared__ float shQ[32 * 257];   // padded rows: conflict-free transposed read
    __shared__ float shR[256];

    const int t   = threadIdx.x;
    const int n0  = blockIdx.x * 32;
    const int b   = n0 >> 10;
    const int hw0 = n0 & 1023;

    if (t < 32) shI[t] = inds[n0 + t];
    __syncthreads();

    // stage 32 gathered fp32 code rows (coalesced 1KB per row)
    for (int r = 0; r < 32; ++r) {
        shQ[r * 257 + t] = emb[shI[r] * DDIM + t];
    }
    __syncthreads();

    const int j    = t & 31;   // row within tile -> contiguous hw
    const int dsub = t >> 5;   // channel group
    float acc = 0.f;
    const int base = (b * DDIM) * 1024 + hw0 + j;
#pragma unroll 4
    for (int dd = 0; dd < 32; ++dd) {
        const int d = dsub * 32 + dd;
        const int g = base + d * 1024;      // coalesced: 32 lanes span hw
        float q  = shQ[j * 257 + d];
        float l  = lat[g];
        float df = q - l;
        acc = __builtin_fmaf(df, df, acc);
        outp[g] = q;                         // straight-through forward == quant
    }

    shR[t] = acc;
    __syncthreads();
    for (int s = 128; s > 0; s >>= 1) {
        if (t < s) shR[t] += shR[t + s];
        __syncthreads();
    }
    if (t == 0) partial[blockIdx.x] = shR[0];
}

// ---------------------------------------------------------------------------
// Kernel 4: deterministic fixed-order final reduction -> two scalars
// ---------------------------------------------------------------------------
__global__ __launch_bounds__(256) void vq_finalize(const float* __restrict__ partial,
                                                   float* __restrict__ outsc) {
    __shared__ float shR[256];
    const int t = threadIdx.x;
    float a = 0.f;
    for (int i = t; i < GATHER_BLOCKS; i += 256) a += partial[i];
    shR[t] = a;
    __syncthreads();
    for (int s = 128; s > 0; s >>= 1) {
        if (t < s) shR[t] += shR[t + s];
        __syncthreads();
    }
    if (t == 0) {
        float mean = shR[0] * (1.0f / 16777216.0f);   // / (N * D)
        outsc[0] = mean;           // embedding_loss
        outsc[1] = 0.25f * mean;   // BETA * commitment_loss (same mean)
    }
}

// ---------------------------------------------------------------------------
extern "C" void kernel_launch(void* const* d_in, const int* in_sizes, int n_in,
                              void* d_out, int out_size, void* d_ws, size_t ws_size,
                              hipStream_t stream) {
    (void)in_sizes; (void)n_in; (void)out_size; (void)ws_size;
    const float* lat = (const float*)d_in[0];   // [64,256,32,32] fp32
    const float* emb = (const float*)d_in[1];   // [1024,256] fp32
    float* outp = (float*)d_out;                // 4194304 tensor + 2 scalars

    char* ws = (char*)d_ws;
    unsigned short* eb    = (unsigned short*)(ws);            // 512 KB bf16 codebook
    float*          enorm = (float*)(ws + 524288);            // 4 KB
    int*            inds  = (int*)(ws + 528384);              // 256 KB
    float*          part  = (float*)(ws + 790528);            // 8 KB

    vq_prep_emb<<<KCODES, 256, 0, stream>>>(emb, eb, enorm);
    vq_argmin<<<NROWS / 128, 128, 0, stream>>>(lat, eb, enorm, inds);
    vq_gather_out<<<GATHER_BLOCKS, 256, 0, stream>>>(lat, emb, inds, outp, part);
    vq_finalize<<<1, 256, 0, stream>>>(part, outp + OUT_ELEMS);
}
